// GatedSelfAttentionModel_67946382623209
// MI455X (gfx1250) — matrix operand order repaired and checked by reading, not compiled, not run on hardware
//
#include <hip/hip_runtime.h>
#include <hip/hip_bf16.h>
#include <math.h>

typedef __attribute__((ext_vector_type(16))) __bf16 v16bf;
typedef __attribute__((ext_vector_type(8)))  float  v8f;

union BF16Frag {
  v16bf v;
  unsigned short us[16];
  uint4 q[2];
};

__device__ __forceinline__ unsigned short f2bf(float f) {
  union { float f; unsigned int u; } c;
  c.f = f;
  unsigned int u = c.u;
  return (unsigned short)((u + 0x7FFFu + ((u >> 16) & 1u)) >> 16); // RNE
}

__device__ __forceinline__ v8f vzero8() {
  return (v8f){0.f, 0.f, 0.f, 0.f, 0.f, 0.f, 0.f, 0.f};
}

// CDNA5 async global->LDS copy: per-lane 16B, GVS addressing (SGPR base +
// per-lane 32-bit byte offset), LDS destination from per-lane VGPR.
// Tracked by ASYNCcnt (CDNA5 ISA).
__device__ __forceinline__ void async_copy_b128(unsigned lds_off, const void* base,
                                                unsigned byte_off) {
  asm volatile("global_load_async_to_lds_b128 %0, %1, %2"
               :
               : "v"(lds_off), "v"(byte_off), "s"(base)
               : "memory");
}

__device__ __forceinline__ void wait_async0() {
  asm volatile("s_wait_asynccnt 0x0" ::: "memory");
}

__device__ __forceinline__ unsigned lds_addr_of(const void* p) {
  // Generic-pointer low 32 bits == LDS byte offset (flat->LDS truncation rule).
  return (unsigned)(size_t)p;
}

// ---------------------------------------------------------------------------
// Generic bf16-WMMA GEMM:  C[M x N] = act(A[M x K] @ W[K x N] + bias)
// Block tile 64x64, 128 threads (4 waves); wave w owns rows 16w..16w+15.
// OMODE 0: fp32 row-major out.  OMODE 2: bf16 out, transposed per (b,h) for V:
//   vt[(b*512 + n_col) * 2048 + token_in_batch]  (n_col = h*64 + d)
// ---------------------------------------------------------------------------
template <bool RELU, int OMODE>
__global__ void __launch_bounds__(128) gemm_bf16(
    const float* __restrict__ A, int lda,
    const float* __restrict__ W,
    const float* __restrict__ bias,
    float* __restrict__ C, int ldc, int col_off,
    int M, int N, int K)
{
  __shared__ __align__(16) unsigned short sA[64 * 40];
  __shared__ __align__(16) unsigned short sB[64 * 40];
  const int tid  = threadIdx.x;
  const int wv   = tid >> 5;
  const int lane = tid & 31;
  const int hf   = lane >> 4;
  const int l16  = lane & 15;
  const int m_blk = blockIdx.y * 64;
  const int n_blk = blockIdx.x * 64;

  v8f acc[4];
#pragma unroll
  for (int t = 0; t < 4; ++t) acc[t] = vzero8();

  for (int k0 = 0; k0 < K; k0 += 32) {
    __syncthreads();
    // A tile: 64 rows x 32 k  (512 float4 chunks / 128 threads)
#pragma unroll
    for (int c = tid; c < 512; c += 128) {
      int m  = c >> 3;
      int kk = (c & 7) << 2;
      const float* src = A + (size_t)(m_blk + m) * lda + (k0 + kk);
      float4 v = *(const float4*)src;
      unsigned short* dst = &sA[m * 40 + kk];
      dst[0] = f2bf(v.x); dst[1] = f2bf(v.y); dst[2] = f2bf(v.z); dst[3] = f2bf(v.w);
    }
    // W tile: 32 k-rows x 64 n, stored transposed [n][k]
#pragma unroll
    for (int c = tid; c < 512; c += 128) {
      int kk = c >> 4;
      int nn = (c & 15) << 2;
      int gk = k0 + kk;
      int gn = n_blk + nn;
      float4 v;
      if (gn + 3 < N) {
        v = *(const float4*)(W + (size_t)gk * N + gn);
      } else {
        v.x = (gn + 0 < N) ? W[(size_t)gk * N + gn + 0] : 0.f;
        v.y = (gn + 1 < N) ? W[(size_t)gk * N + gn + 1] : 0.f;
        v.z = (gn + 2 < N) ? W[(size_t)gk * N + gn + 2] : 0.f;
        v.w = (gn + 3 < N) ? W[(size_t)gk * N + gn + 3] : 0.f;
      }
      sB[(nn + 0) * 40 + kk] = f2bf(v.x);
      sB[(nn + 1) * 40 + kk] = f2bf(v.y);
      sB[(nn + 2) * 40 + kk] = f2bf(v.z);
      sB[(nn + 3) * 40 + kk] = f2bf(v.w);
    }
    __syncthreads();

    // A fragment: lane l16 = row, element e -> k = (e/8)*16 + hf*8 + e%8
    BF16Frag afr;
    {
      const unsigned short* ap = &sA[(wv * 16 + l16) * 40 + hf * 8];
      afr.q[0] = *(const uint4*)ap;
      afr.q[1] = *(const uint4*)(ap + 16);
    }
#pragma unroll
    for (int nt = 0; nt < 4; ++nt) {
      // B fragment: lane l16 = col, element e -> k = (e/8)*16 + hf*8 + e%8
      BF16Frag bfr;
      const unsigned short* bp = &sB[(nt * 16 + l16) * 40 + hf * 8];
      bfr.q[0] = *(const uint4*)bp;
      bfr.q[1] = *(const uint4*)(bp + 16);
      acc[nt] = __builtin_amdgcn_wmma_f32_16x16x32_bf16(
          false, afr.v, false, bfr.v, (short)0, acc[nt], false, false);
    }
  }

  // Epilogue: C/D layout -> lane l16 = col, VGPR r -> row r + 8*hf
#pragma unroll
  for (int nt = 0; nt < 4; ++nt) {
    int gn = n_blk + nt * 16 + l16;
    if (gn >= N) continue;
    float bv = bias ? bias[gn] : 0.f;
#pragma unroll
    for (int r = 0; r < 8; ++r) {
      int gm = m_blk + wv * 16 + hf * 8 + r;
      float val = acc[nt][r] + bv;
      if (RELU) val = fmaxf(val, 0.f);
      if (OMODE == 2) {
        int bb = gm >> 11;             // batch
        int n  = gm & 2047;            // token in batch
        ((unsigned short*)C)[((size_t)(bb * 512 + gn)) * 2048 + n] = f2bf(val);
      } else {
        C[(size_t)gm * ldc + col_off + gn] = val;
      }
    }
  }
}

// ---------------------------------------------------------------------------
// Flash attention: one block = one (b,h) and a 64-row Q tile; 4 waves,
// each wave owns 16 query rows. Q/K/V are bf16 in global; K and V^T tiles are
// staged by GLOBAL_LOAD_ASYNC_TO_LDS_B128 (no VGPR round-trip), Q fragments
// load straight from global as b128s. bf16 WMMA for QK^T and PV, fp32 softmax.
// ---------------------------------------------------------------------------
__global__ void __launch_bounds__(128) attn_fa(
    const unsigned short* __restrict__ Qn, const unsigned short* __restrict__ Kn,
    const unsigned short* __restrict__ Vt, const float* __restrict__ qfull,
    float* __restrict__ AO)
{
  __shared__ __align__(16) unsigned short sK[64 * 72];      // [key][d], 144B rows
  __shared__ __align__(16) unsigned short sV[64 * 72];      // [d][key], 144B rows
  __shared__ __align__(16) unsigned short sP[4][16 * 72];   // per-wave P staging

  const int bh = blockIdx.y;
  const int b  = bh >> 3;
  const int hh = bh & 7;
  const int q0 = blockIdx.x * 64;
  const int tid  = threadIdx.x;
  const int wv   = tid >> 5;
  const int lane = tid & 31;
  const int hf   = lane >> 4;
  const int l16  = lane & 15;

  // Q fragments in A layout directly from global bf16:
  // element e of frag f -> d = f*32 + (e/8)*16 + hf*8 + e%8  (two 16B runs)
  BF16Frag qfr[2];
  {
    const unsigned short* qp =
        Qn + (size_t)(b * 2048 + q0 + wv * 16 + l16) * 512 + hh * 64;
#pragma unroll
    for (int f = 0; f < 2; ++f) {
      qfr[f].q[0] = *(const uint4*)(qp + f * 32 + hf * 8);
      qfr[f].q[1] = *(const uint4*)(qp + f * 32 + hf * 8 + 16);
    }
  }

  const unsigned ldsK0 = lds_addr_of(&sK[0]);
  const unsigned ldsV0 = lds_addr_of(&sV[0]);

  float mrow[8], lrow[8];
  v8f oacc[4];
#pragma unroll
  for (int r = 0; r < 8; ++r) { mrow[r] = -1e30f; lrow[r] = 0.f; }
#pragma unroll
  for (int t = 0; t < 4; ++t) oacc[t] = vzero8();

  for (int j0 = 0; j0 < 2048; j0 += 64) {
    __syncthreads();  // previous iteration's LDS reads complete before overwrite
    // Async-stage K tile [key][d] and V^T tile [d][key]: 64 rows x 128B each,
    // 8 chunks/row, 512 chunks / 128 threads = 4 per thread per tensor.
#pragma unroll
    for (int i = 0; i < 4; ++i) {
      int cid = tid + i * 128;
      int row = cid >> 3;           // key (for K) / d (for V^T)
      int ch  = cid & 7;
      unsigned ldst = (unsigned)(row * 144 + ch * 16);
      unsigned kgo = (unsigned)((((b * 2048 + j0 + row) * 512) + hh * 64 + ch * 8) * 2);
      async_copy_b128(ldsK0 + ldst, Kn, kgo);
      unsigned vgo = (unsigned)((((bh * 64 + row) * 2048) + j0 + ch * 8) * 2);
      async_copy_b128(ldsV0 + ldst, Vt, vgo);
    }
    wait_async0();
    __syncthreads();

    // S = Q K^T : 4 tiles of 16 keys, contraction over d (2 WMMAs each)
    v8f sacc[4];
#pragma unroll
    for (int nt = 0; nt < 4; ++nt) {
      sacc[nt] = vzero8();
#pragma unroll
      for (int f = 0; f < 2; ++f) {
        BF16Frag kf;
        const unsigned short* bp = &sK[(nt * 16 + l16) * 72 + f * 32 + hf * 8];
        kf.q[0] = *(const uint4*)bp;
        kf.q[1] = *(const uint4*)(bp + 16);
        sacc[nt] = __builtin_amdgcn_wmma_f32_16x16x32_bf16(
            false, qfr[f].v, false, kf.v, (short)0, sacc[nt], false, false);
      }
    }

    // Online softmax (row = r + 8*hf; reduce across the 16 lanes of each half)
    float alpha[8], psum[8];
#pragma unroll
    for (int r = 0; r < 8; ++r) {
      float mx = fmaxf(fmaxf(sacc[0][r], sacc[1][r]), fmaxf(sacc[2][r], sacc[3][r]));
#pragma unroll
      for (int s = 1; s < 16; s <<= 1) mx = fmaxf(mx, __shfl_xor(mx, s, 32));
      float mnew = fmaxf(mrow[r], mx);
      alpha[r] = __expf(mrow[r] - mnew);
      mrow[r]  = mnew;
      psum[r]  = 0.f;
    }
#pragma unroll
    for (int nt = 0; nt < 4; ++nt)
#pragma unroll
      for (int r = 0; r < 8; ++r) {
        float p = __expf(sacc[nt][r] - mrow[r]);
        sacc[nt][r] = p;
        psum[r] += p;
      }
#pragma unroll
    for (int r = 0; r < 8; ++r) {
#pragma unroll
      for (int s = 1; s < 16; s <<= 1) psum[r] += __shfl_xor(psum[r], s, 32);
      lrow[r] = lrow[r] * alpha[r] + psum[r];
#pragma unroll
      for (int nt = 0; nt < 4; ++nt) oacc[nt][r] *= alpha[r];
    }

    // Restage P through per-wave LDS to convert C-layout -> A-layout (bf16)
    unsigned short* sp = &sP[wv][0];
#pragma unroll
    for (int nt = 0; nt < 4; ++nt)
#pragma unroll
      for (int r = 0; r < 8; ++r)
        sp[(hf * 8 + r) * 72 + nt * 16 + l16] = f2bf(sacc[nt][r]);
    __syncthreads();

    // O += P V : contraction over 64 keys (2 frags), 4 d-tiles
#pragma unroll
    for (int f = 0; f < 2; ++f) {
      BF16Frag pfr;
      const unsigned short* pp = &sP[wv][l16 * 72 + f * 32 + hf * 8];
      pfr.q[0] = *(const uint4*)pp;
      pfr.q[1] = *(const uint4*)(pp + 16);
#pragma unroll
      for (int dt = 0; dt < 4; ++dt) {
        BF16Frag vf;
        const unsigned short* vp = &sV[(dt * 16 + l16) * 72 + f * 32 + hf * 8];
        vf.q[0] = *(const uint4*)vp;
        vf.q[1] = *(const uint4*)(vp + 16);
        oacc[dt] = __builtin_amdgcn_wmma_f32_16x16x32_bf16(
            false, pfr.v, false, vf.v, (short)0, oacc[dt], false, false);
      }
    }
  }

  // Epilogue: 1/l normalization + sigmoid(gate) and store
#pragma unroll
  for (int r = 0; r < 8; ++r) {
    size_t token = (size_t)(b * 2048 + q0 + wv * 16 + hf * 8 + r);
    float g  = qfull[token * 520 + 512 + hh];
    float sg = 1.0f / (1.0f + __expf(-g));
    float scl = sg / lrow[r];
#pragma unroll
    for (int dt = 0; dt < 4; ++dt)
      AO[token * 512 + hh * 64 + dt * 16 + l16] = oacc[dt][r] * scl;
  }
}

// ---------------------------------------------------------------------------
// Small VALU kernels (tiny-K work where WMMA would be wasted)
// ---------------------------------------------------------------------------

// Per-(token,head) RMSNorm over 64 dims -> bf16 output; one warp per pair.
__global__ void __launch_bounds__(128) rms_head(
    const float* __restrict__ X, int ldx,
    const float* __restrict__ w, float scale, unsigned short* __restrict__ Y)
{
  int idx  = blockIdx.x * 4 + (threadIdx.x >> 5);
  int lane = threadIdx.x & 31;
  int token = idx >> 3;
  int hh    = idx & 7;
  const float* src = X + (size_t)token * ldx + hh * 64;
  float a = src[lane];
  float b = src[lane + 32];
  float ss = a * a + b * b;
#pragma unroll
  for (int s = 1; s < 32; s <<= 1) ss += __shfl_xor(ss, s, 32);
  float r = rsqrtf(ss * (1.0f / 64.0f) + 1e-6f);
  unsigned short* dst = Y + (size_t)token * 512 + hh * 64;
  dst[lane]      = f2bf(a * r * w[lane] * scale);
  dst[lane + 32] = f2bf(b * r * w[lane + 32] * scale);
}

// Position MLP 4 -> 64 -> 128, writes cat[:, 640:768]
__global__ void __launch_bounds__(128) pos_mlp(
    const float* __restrict__ posf,
    const float* __restrict__ w1, const float* __restrict__ b1,
    const float* __restrict__ w2, const float* __restrict__ b2,
    float* __restrict__ cat)
{
  __shared__ float h1[64];
  __shared__ float pf[4];
  int token = blockIdx.x;
  int tid = threadIdx.x;
  if (tid == 0) {
    float x = posf[token * 3 + 0], y = posf[token * 3 + 1], z = posf[token * 3 + 2];
    float n = sqrtf(x * x + y * y + z * z);
    float inv = 1.0f / (n + 1e-7f);
    pf[0] = x * inv; pf[1] = y * inv; pf[2] = z * inv; pf[3] = n;
  }
  __syncthreads();
  if (tid < 64) {
    float s = b1[tid];
#pragma unroll
    for (int i = 0; i < 4; ++i) s += pf[i] * w1[i * 64 + tid];
    h1[tid] = fmaxf(s, 0.f);
  }
  __syncthreads();
  float s = b2[tid];
  for (int i = 0; i < 64; ++i) s += h1[i] * w2[i * 128 + tid];
  cat[(size_t)token * 768 + 640 + tid] = s;
}

// tdt = diffusion_time @ yy_w + yy_b  (4x128 @ 128x128), dual write
__global__ void __launch_bounds__(128) time_proj(
    const float* __restrict__ dt, const float* __restrict__ w,
    const float* __restrict__ bias, float* __restrict__ tdt,
    float* __restrict__ out_tdt)
{
  int j = threadIdx.x;
  for (int b = 0; b < 4; ++b) {
    float s = bias[j];
    for (int i = 0; i < 128; ++i) s += dt[b * 128 + i] * w[i * 128 + j];
    tdt[b * 128 + j] = s;
    out_tdt[b * 128 + j] = s;
  }
}

// Broadcast tdt into cat[:, 512:640]
__global__ void __launch_bounds__(128) time_bcast(
    const float* __restrict__ tdt, float* __restrict__ cat)
{
  int token = blockIdx.x;
  int j = threadIdx.x;
  int b = token >> 11;
  cat[(size_t)token * 768 + 512 + j] = tdt[b * 128 + j];
}

// pos_out = out @ hp_w + hp_b  (N=3: plain dot products)
__global__ void __launch_bounds__(128) pos_head(
    const float* __restrict__ outb, const float* __restrict__ w,
    const float* __restrict__ bias, float* __restrict__ po)
{
  int idx = blockIdx.x * 128 + threadIdx.x;
  if (idx >= 8192 * 3) return;
  int token = idx / 3;
  int o = idx % 3;
  float s = bias[o];
  const float* x = outb + (size_t)token * 512;
  for (int i = 0; i < 512; ++i) s += x[i] * w[i * 3 + o];
  po[idx] = s;
}

// ---------------------------------------------------------------------------
extern "C" void kernel_launch(void* const* d_in, const int* in_sizes, int n_in,
                              void* d_out, int out_size, void* d_ws, size_t ws_size,
                              hipStream_t stream)
{
  (void)in_sizes; (void)n_in; (void)out_size; (void)ws_size;
  const float* expr_x  = (const float*)d_in[0];
  const float* difft   = (const float*)d_in[1];
  const float* posf    = (const float*)d_in[2];
  const float* pos_w1  = (const float*)d_in[3];
  const float* pos_b1  = (const float*)d_in[4];
  const float* pos_w2  = (const float*)d_in[5];
  const float* pos_b2  = (const float*)d_in[6];
  const float* expr_w1 = (const float*)d_in[7];
  const float* expr_b1 = (const float*)d_in[8];
  const float* expr_w2 = (const float*)d_in[9];
  const float* expr_b2 = (const float*)d_in[10];
  const float* yy_w    = (const float*)d_in[11];
  const float* yy_b    = (const float*)d_in[12];
  const float* cat_w   = (const float*)d_in[13];
  const float* cat_b   = (const float*)d_in[14];
  const float* q_w     = (const float*)d_in[15];
  const float* q_b     = (const float*)d_in[16];
  const float* k_w     = (const float*)d_in[17];
  const float* k_b     = (const float*)d_in[18];
  const float* v_w     = (const float*)d_in[19];
  const float* v_b     = (const float*)d_in[20];
  const float* o_w     = (const float*)d_in[21];
  const float* o_b     = (const float*)d_in[22];
  const float* qn_w    = (const float*)d_in[23];
  const float* kn_w    = (const float*)d_in[24];
  const float* hp_w    = (const float*)d_in[25];
  const float* hp_b    = (const float*)d_in[26];
  const float* he_w    = (const float*)d_in[27];
  const float* he_b    = (const float*)d_in[28];

  float* ws    = (float*)d_ws;
  float* tmp1  = ws;                       // 8192*64
  float* cat   = ws + 524288;              // 8192*768
  float* hbuf  = ws + 6815744;             // 8192*512  (ao reuses)
  float* qfull = ws + 11010048;            // 8192*520
  float* kbuf  = ws + 15269888;            // 8192*512  (out reuses)
  unsigned short* vt  = (unsigned short*)(ws + 19464192); // 8192*512 bf16 (V^T)
  unsigned short* qnb = (unsigned short*)(ws + 21561344); // 8192*512 bf16
  unsigned short* knb = (unsigned short*)(ws + 23658496); // 8192*512 bf16
  float* tdt   = ws + 25755648;            // 512
  float* ao    = hbuf;
  float* outb  = kbuf;

  float* out_expr = (float*)d_out;            // 8192*512
  float* out_pos  = out_expr + 8192 * 512;    // 8192*3
  float* out_tdt  = out_pos + 8192 * 3;       // 4*128

  dim3 blk(128);
  // expr MLP
  gemm_bf16<true , 0><<<dim3(1, 128), blk, 0, stream>>>(expr_x, 512, expr_w1, expr_b1, tmp1, 64, 0, 8192, 64, 512);
  gemm_bf16<false, 0><<<dim3(8, 128), blk, 0, stream>>>(tmp1, 64, expr_w2, expr_b2, cat, 768, 0, 8192, 512, 64);
  // time projection + broadcast, position MLP -> cat columns
  time_proj<<<dim3(1), blk, 0, stream>>>(difft, yy_w, yy_b, tdt, out_tdt);
  time_bcast<<<dim3(8192), blk, 0, stream>>>(tdt, cat);
  pos_mlp<<<dim3(8192), blk, 0, stream>>>(posf, pos_w1, pos_b1, pos_w2, pos_b2, cat);
  // h = cat @ cat_w
  gemm_bf16<false, 0><<<dim3(8, 128), blk, 0, stream>>>(cat, 768, cat_w, cat_b, hbuf, 512, 0, 8192, 512, 768);
  // qfull (Q + gate), K fp32; V straight to bf16 transposed-per-head layout
  gemm_bf16<false, 0><<<dim3(9, 128), blk, 0, stream>>>(hbuf, 512, q_w, q_b, qfull, 520, 0, 8192, 520, 512);
  gemm_bf16<false, 0><<<dim3(8, 128), blk, 0, stream>>>(hbuf, 512, k_w, k_b, kbuf, 512, 0, 8192, 512, 512);
  gemm_bf16<false, 2><<<dim3(8, 128), blk, 0, stream>>>(hbuf, 512, v_w, v_b, (float*)vt, 512, 0, 8192, 512, 512);
  // RMSNorm -> bf16 (fold 1/sqrt(64) into Q)
  rms_head<<<dim3(16384), blk, 0, stream>>>(qfull, 520, qn_w, 0.125f, qnb);
  rms_head<<<dim3(16384), blk, 0, stream>>>(kbuf, 512, kn_w, 1.0f, knb);
  // gated flash attention with async LDS staging
  attn_fa<<<dim3(32, 32), blk, 0, stream>>>(qnb, knb, vt, qfull, ao);
  // output projection + heads
  gemm_bf16<false, 0><<<dim3(8, 128), blk, 0, stream>>>(ao, 512, o_w, o_b, outb, 512, 0, 8192, 512, 512);
  gemm_bf16<false, 0><<<dim3(8, 128), blk, 0, stream>>>(outb, 512, he_w, he_b, out_expr, 512, 0, 8192, 512, 512);
  pos_head<<<dim3(192), blk, 0, stream>>>(outb, hp_w, hp_b, out_pos);
}
